// UltraOptimizedINLBlock_970662609167
// MI455X (gfx1250) — compile-verified, matrix operand order checked by reading, not run on hardware
//
#include <hip/hip_runtime.h>
#include <hip/hip_bf16.h>
#include <cstdint>

// ---------------------------------------------------------------------------
// Problem constants (fixed by the reference)
// ---------------------------------------------------------------------------
constexpr int kB  = 2;
constexpr int kS  = 2048;
constexpr int kD  = 1024;
constexpr int kH  = 16;
constexpr int kHD = 64;
constexpr int kR  = kB * kS;      // 4096 rows
constexpr int kN3 = 3 * kD;       // 3072
constexpr int kF  = 4096;         // FF_DIM
constexpr float kDT = 0.1f;

typedef _Float16 h8  __attribute__((ext_vector_type(8)));
typedef _Float16 h16 __attribute__((ext_vector_type(16)));
typedef float    f8  __attribute__((ext_vector_type(8)));

// ---------------------------------------------------------------------------
// WMMA helper (CDNA5 16x16x32 f16 -> f32)
// ---------------------------------------------------------------------------
__device__ __forceinline__ f8 wmma_f16(h16 a, h16 b, f8 c) {
  // 8 args: (neg_a, A, neg_b, B, c_mod, C, reuse_a, reuse_b)
  return __builtin_amdgcn_wmma_f32_16x16x32_f16(false, a, false, b, (short)0, c,
                                                false, false);
}

// ---------------------------------------------------------------------------
// Async global->LDS 16-byte copy (CDNA5 ASYNCcnt path).  The LDS operand is
// the wave-relative LDS byte address = low 32 bits of the generic shared
// pointer.
// ---------------------------------------------------------------------------
__device__ __forceinline__ void async_cp16(const _Float16* g, _Float16* l) {
#if defined(__gfx1250__)
  const unsigned lds = (unsigned)(uintptr_t)l;
  asm volatile("global_load_async_to_lds_b128 %0, %1, off"
               :: "v"(lds), "v"(g) : "memory");
#else
  *(h8*)l = *(const h8*)g;
#endif
}

__device__ __forceinline__ void async_wait0() {
#if defined(__gfx1250__)
  asm volatile("s_wait_asynccnt 0" ::: "memory");
#endif
}

// Load one 16x32 f16 A/B fragment from row-major storage (LDS or global).
// A-operand layout (ISA 7.12.2): lane L holds row (L&15); K-chunks
// [base, base+8) and [base+16, base+24) with base = (L<16 ? 0 : 8).
// B-operand is the mirror (lane = N column, same K chunking), so the same
// loader is used with "rows" = the N index of W / K / V^T.
__device__ __forceinline__ h16 frag_ld(const _Float16* base, int row0, int col0,
                                       int stride) {
  const int lane = threadIdx.x & 31;
  const int r = row0 + (lane & 15);
  const int c = col0 + ((lane & 16) ? 8 : 0);
  const _Float16* p = base + (size_t)r * stride + c;
  h8 lo = *(const h8*)(p);
  h8 hi = *(const h8*)(p + 16);
  h16 f;
#pragma unroll
  for (int i = 0; i < 8; ++i) { f[i] = lo[i]; f[i + 8] = hi[i]; }
  return f;
}

// ---------------------------------------------------------------------------
// Generic f16 GEMM:  C[M,N] = A[M,K] * Wt[N,K]^T + bias[N]  (+ fused epilogue)
//   MODE 0: store f32                  (ctrl projection)
//   MODE 1: QKV scatter to head-major  (q -> o16, k -> aux16a, v^T -> aux16b)
//   MODE 2: out-proj: o32=context f32, o16=context f16, aux32 = resid + C
//   MODE 3: GELU -> f16                (FF1)
//   MODE 4: o32 = resid + C            (FF2 -> d_out)
// Block: 128 threads = 4 waves; each wave owns a 32x32 quadrant of a 64x64
// C tile.  K stepped by 32 with double-buffered async global->LDS staging:
// tile k+32 DMAs into the alternate LDS buffer while WMMAs consume tile k.
// ---------------------------------------------------------------------------
template <int MODE>
__global__ __launch_bounds__(128) void gemm_kernel(
    const _Float16* __restrict__ A, const _Float16* __restrict__ Wt,
    const float* __restrict__ bias, int M, int N, int K,
    float* __restrict__ o32, _Float16* __restrict__ o16,
    const float* __restrict__ resid, float* __restrict__ aux32,
    _Float16* __restrict__ aux16a, _Float16* __restrict__ aux16b) {
  __shared__ _Float16 lA[2][64 * 40];
  __shared__ _Float16 lB[2][64 * 40];
  const int tid  = threadIdx.x;
  const int lane = tid & 31;
  const int wave = tid >> 5;
  const int wm = wave & 1, wn = wave >> 1;
  const int m0 = blockIdx.y * 64;
  const int n0 = blockIdx.x * 64;

  // Stage one 64x32 A tile + 64x32 B tile into LDS buffer `buf` (async DMA).
  auto stage = [&](int k0, int buf) {
#pragma unroll
    for (int pass = 0; pass < 2; ++pass) {
      const int c   = tid + pass * 128;   // 0..255
      const int row = c >> 2;             // 0..63
      const int col = (c & 3) * 8;        // 0,8,16,24
      async_cp16(&A[(size_t)(m0 + row) * K + k0 + col],
                 &lA[buf][row * 40 + col]);
      async_cp16(&Wt[(size_t)(n0 + row) * K + k0 + col],
                 &lB[buf][row * 40 + col]);
      if (k0 + 32 < K) {  // keep L2 one tile ahead of the DMA
        __builtin_prefetch(&A[(size_t)(m0 + row) * K + k0 + 32 + col], 0, 3);
        __builtin_prefetch(&Wt[(size_t)(n0 + row) * K + k0 + 32 + col], 0, 3);
      }
    }
  };

  f8 acc[2][2] = {};

  stage(0, 0);
  async_wait0();
  __syncthreads();

  int cur = 0;
  for (int k0 = 0; k0 < K; k0 += 32) {
    if (k0 + 32 < K) stage(k0 + 32, cur ^ 1);  // overlap DMA with WMMAs

    h16 a0 = frag_ld(lA[cur], wm * 32, 0, 40);
    h16 a1 = frag_ld(lA[cur], wm * 32 + 16, 0, 40);
    h16 b0 = frag_ld(lB[cur], wn * 32, 0, 40);
    h16 b1 = frag_ld(lB[cur], wn * 32 + 16, 0, 40);
    acc[0][0] = wmma_f16(a0, b0, acc[0][0]);
    acc[0][1] = wmma_f16(a0, b1, acc[0][1]);
    acc[1][0] = wmma_f16(a1, b0, acc[1][0]);
    acc[1][1] = wmma_f16(a1, b1, acc[1][1]);

    async_wait0();     // my DMA for tile k+32 has landed
    __syncthreads();   // everyone's has landed; frag reads above are done
    cur ^= 1;
  }

  const int hi8 = (lane >> 4) << 3;
  const int nl  = lane & 15;
#pragma unroll
  for (int i = 0; i < 2; ++i) {
#pragma unroll
    for (int j = 0; j < 2; ++j) {
      const int mb = m0 + wm * 32 + i * 16;
      const int nb = n0 + wn * 32 + j * 16;
#pragma unroll
      for (int r = 0; r < 8; ++r) {
        const int m = mb + r + hi8;       // C layout: m = r + 8*(lane>=16)
        const int n = nb + nl;            //           n = lane & 15
        float v = acc[i][j][r] + bias[n];
        const size_t idx = (size_t)m * N + n;
        if constexpr (MODE == 0) {
          o32[idx] = v;
        } else if constexpr (MODE == 1) {
          const int b = m >> 11, s = m & 2047;
          const int part = n >> 10, id = n & 1023;
          const int h = id >> 6, d = id & 63;
          if (part == 0)
            o16[((((size_t)b * kH + h) * kS) + s) * kHD + d] = (_Float16)v;
          else if (part == 1)
            aux16a[((((size_t)b * kH + h) * kS) + s) * kHD + d] = (_Float16)v;
          else  // V stored transposed: [b,h,d,s] for the P*V B-operand
            aux16b[((((size_t)b * kH + h) * kHD) + d) * kS + s] = (_Float16)v;
        } else if constexpr (MODE == 2) {
          o32[idx] = v;                       // context (fp32, for drive)
          o16[idx] = (_Float16)v;             // context (f16, ctrl GEMM input)
          aux32[idx] = resid[idx] + v;        // x1 = x + attn_output
        } else if constexpr (MODE == 3) {
          float g = 0.5f * v *
                    (1.f + tanhf(0.7978845608f * (v + 0.044715f * v * v * v)));
          o16[idx] = (_Float16)g;
        } else {  // MODE == 4
          o32[idx] = resid[idx] + v;
        }
      }
    }
  }
}

// ---------------------------------------------------------------------------
// Flash attention (causal), f16 WMMA, online softmax.
// Grid: (S/64, H, B); 128 threads = 4 waves; each wave owns 16 query rows.
// Per 32-key step: 4 WMMAs for S = Q*K^T, per-lane softmax on the C layout
// (width-16 shfl_xor row reductions), P staged via wave-private LDS, then
// 4 WMMAs for P*V (V stored transposed [b,h,d,s]).
// ---------------------------------------------------------------------------
__global__ __launch_bounds__(128) void flash_attn_kernel(
    const _Float16* __restrict__ Q, const _Float16* __restrict__ Kh,
    const _Float16* __restrict__ Vt, _Float16* __restrict__ ctx) {
  __shared__ _Float16 lP[4][16 * 40];
  const int tid = threadIdx.x, lane = tid & 31, wave = tid >> 5;
  const int h = blockIdx.y, b = blockIdx.z;
  const int q0 = blockIdx.x * 64 + wave * 16;
  const _Float16* Qb = Q  + (((size_t)b * kH + h) * kS) * kHD;
  const _Float16* Kb = Kh + (((size_t)b * kH + h) * kS) * kHD;
  const _Float16* Vb = Vt + (((size_t)b * kH + h) * kHD) * kS;
  _Float16* lPw = lP[wave];

  const h16 qa0 = frag_ld(Qb, q0, 0, kHD);
  const h16 qa1 = frag_ld(Qb, q0, 32, kHD);

  f8 acc[4] = {};
  float Mrow[8], Lrow[8];
#pragma unroll
  for (int r = 0; r < 8; ++r) { Mrow[r] = -1e30f; Lrow[r] = 0.f; }

  const int hi8 = (lane >> 4) << 3;
  const int nl  = lane & 15;
  const int ntiles = (q0 + 15) / 32 + 1;  // causal: keys <= q0+15

  for (int t = 0; t < ntiles; ++t) {
    const int k0 = t * 32;
    f8 S0 = {}, S1 = {};
    {
      h16 bk00 = frag_ld(Kb, k0, 0, kHD);
      h16 bk01 = frag_ld(Kb, k0, 32, kHD);
      S0 = wmma_f16(qa0, bk00, S0);
      S0 = wmma_f16(qa1, bk01, S0);
      h16 bk10 = frag_ld(Kb, k0 + 16, 0, kHD);
      h16 bk11 = frag_ld(Kb, k0 + 16, 32, kHD);
      S1 = wmma_f16(qa0, bk10, S1);
      S1 = wmma_f16(qa1, bk11, S1);
    }

    float s0[8], s1[8], tmax[8];
#pragma unroll
    for (int r = 0; r < 8; ++r) {
      const int m = q0 + r + hi8;
      s0[r] = (k0 + nl      <= m) ? S0[r] * 0.125f : -1e30f;  // 1/sqrt(64)
      s1[r] = (k0 + 16 + nl <= m) ? S1[r] * 0.125f : -1e30f;
      tmax[r] = fmaxf(s0[r], s1[r]);
    }
    for (int off = 8; off >= 1; off >>= 1) {
#pragma unroll
      for (int r = 0; r < 8; ++r)
        tmax[r] = fmaxf(tmax[r], __shfl_xor(tmax[r], off, 16));
    }

    float rs[8];
#pragma unroll
    for (int r = 0; r < 8; ++r) {
      const float Mn = fmaxf(Mrow[r], tmax[r]);
      const float sc = __expf(Mrow[r] - Mn);
      Mrow[r] = Mn;
      const float p0 = __expf(s0[r] - Mn);
      const float p1 = __expf(s1[r] - Mn);
      rs[r] = p0 + p1;
      Lrow[r] *= sc;
      acc[0][r] *= sc; acc[1][r] *= sc; acc[2][r] *= sc; acc[3][r] *= sc;
      lPw[(r + hi8) * 40 + nl]      = (_Float16)p0;
      lPw[(r + hi8) * 40 + 16 + nl] = (_Float16)p1;
    }
    for (int off = 8; off >= 1; off >>= 1) {
#pragma unroll
      for (int r = 0; r < 8; ++r) rs[r] += __shfl_xor(rs[r], off, 16);
    }
#pragma unroll
    for (int r = 0; r < 8; ++r) Lrow[r] += rs[r];

    // P * V : P is the wave's 16x32 LDS tile; B-operand rows are V^T rows (d)
    const h16 pa = frag_ld(lPw, 0, 0, 40);
#pragma unroll
    for (int c = 0; c < 4; ++c) {
      h16 bv = frag_ld(Vb, c * 16, k0, kS);
      acc[c] = wmma_f16(pa, bv, acc[c]);
    }
  }

  const size_t rowbase = (size_t)b * kS;
#pragma unroll
  for (int c = 0; c < 4; ++c) {
#pragma unroll
    for (int r = 0; r < 8; ++r) {
      const int m = q0 + r + hi8;
      const int col = h * 64 + c * 16 + nl;
      ctx[(rowbase + m) * kD + col] = (_Float16)(acc[c][r] / Lrow[r]);
    }
  }
}

// ---------------------------------------------------------------------------
// LayerNorm over D=1024: one 256-thread block per row, 4 elems/thread.
// ---------------------------------------------------------------------------
__global__ __launch_bounds__(256) void layernorm_kernel(
    const float* __restrict__ x, const float* __restrict__ g,
    const float* __restrict__ bta, float* __restrict__ o32,
    _Float16* __restrict__ o16) {
  __shared__ float ssum[256];
  __shared__ float ssq[256];
  const int row = blockIdx.x, tid = threadIdx.x;
  const float4 v4 = *(const float4*)&x[(size_t)row * kD + tid * 4];
  float vv[4] = {v4.x, v4.y, v4.z, v4.w};
  float s = vv[0] + vv[1] + vv[2] + vv[3];
  float q = vv[0] * vv[0] + vv[1] * vv[1] + vv[2] * vv[2] + vv[3] * vv[3];
  ssum[tid] = s; ssq[tid] = q;
  __syncthreads();
  for (int o = 128; o > 0; o >>= 1) {
    if (tid < o) { ssum[tid] += ssum[tid + o]; ssq[tid] += ssq[tid + o]; }
    __syncthreads();
  }
  const float mean = ssum[0] * (1.f / kD);
  const float var  = ssq[0] * (1.f / kD) - mean * mean;
  const float rstd = rsqrtf(var + 1e-5f);
#pragma unroll
  for (int i = 0; i < 4; ++i) {
    const int c = tid * 4 + i;
    const float val = (vv[i] - mean) * rstd * g[c] + bta[c];
    if (o32) o32[(size_t)row * kD + c] = val;
    if (o16) o16[(size_t)row * kD + c] = (_Float16)val;
  }
}

// ---------------------------------------------------------------------------
// INL equilibrium loop: fully elementwise; writes both trajectories to d_out
// and x2 = x1 + x_state to workspace.
// ---------------------------------------------------------------------------
__global__ __launch_bounds__(256) void inl_kernel(
    const float* __restrict__ xnorm, const float* __restrict__ ctrl,
    const float* __restrict__ context, const float* __restrict__ mu_g,
    const float* __restrict__ mu_off, const float* __restrict__ x1,
    float* __restrict__ x2, float* __restrict__ tx, float* __restrict__ tv) {
  const size_t i = (size_t)blockIdx.x * 256 + threadIdx.x;  // < kR*kD
  const int row = (int)(i >> 10);
  const int col = (int)(i & 1023);
  const size_t cb = (size_t)row * kN3 + col;
  const float a  = ctrl[cb];
  const float bb = ctrl[cb + 1024];
  const float gg = ctrl[cb + 2048];
  const float alpha = 1.f / (1.f + __expf(-a));
  const float beta  = (bb > 20.f) ? bb : log1pf(__expf(bb));
  const float gate  = 1.f / (1.f + __expf(-gg));
  const float drive = gate * context[i];
  const float mu = mu_g[col >> 6] + mu_off[col];
  float xs = xnorm[i], vs = 0.f;
  const size_t tb = ((size_t)row * 9) * kD + col;
  tx[tb] = xs;
  tv[tb] = 0.f;
#pragma unroll
  for (int t = 1; t <= 8; ++t) {
    vs += kDT * (alpha * (mu - xs) - beta * vs + drive);
    xs += kDT * vs;
    tx[tb + (size_t)t * kD] = xs;
    tv[tb + (size_t)t * kD] = vs;
  }
  x2[i] = x1[i] + xs;
}

__global__ __launch_bounds__(256) void f32_to_f16_kernel(
    const float* __restrict__ in, _Float16* __restrict__ out, int n) {
  const int i = blockIdx.x * 256 + threadIdx.x;
  if (i < n) out[i] = (_Float16)in[i];
}

// ---------------------------------------------------------------------------
// Launch
// ---------------------------------------------------------------------------
extern "C" void kernel_launch(void* const* d_in, const int* in_sizes, int n_in,
                              void* d_out, int out_size, void* d_ws,
                              size_t ws_size, hipStream_t stream) {
  (void)in_sizes; (void)n_in; (void)out_size; (void)ws_size;
  const float* x         = (const float*)d_in[0];
  const float* g_attn    = (const float*)d_in[1];
  const float* b_attn    = (const float*)d_in[2];
  const float* g1        = (const float*)d_in[3];
  const float* b1        = (const float*)d_in[4];
  const float* g2        = (const float*)d_in[5];
  const float* b2        = (const float*)d_in[6];
  const float* in_proj_w = (const float*)d_in[7];
  const float* in_proj_b = (const float*)d_in[8];
  const float* out_proj_w= (const float*)d_in[9];
  const float* out_proj_b= (const float*)d_in[10];
  const float* ctrl_w    = (const float*)d_in[11];
  const float* ctrl_b    = (const float*)d_in[12];
  const float* mu_global = (const float*)d_in[13];
  const float* mu_offsets= (const float*)d_in[14];
  const float* w_ff1     = (const float*)d_in[15];
  const float* b_ff1     = (const float*)d_in[16];
  const float* w_ff2     = (const float*)d_in[17];
  const float* b_ff2     = (const float*)d_in[18];

  float* out_x  = (float*)d_out;
  float* trajx  = out_x + (size_t)kR * kD;
  float* trajv  = trajx + (size_t)kR * 9 * kD;

  // ---- workspace carve-up (all sizes are multiples of 256 B) ----
  char* p = (char*)d_ws;
  auto take = [&](size_t bytes) { char* q = p; p += bytes; return q; };
  _Float16* xn_h      = (_Float16*)take((size_t)kR * kD * 2);
  _Float16* wqkv_h    = (_Float16*)take((size_t)kN3 * kD * 2);
  _Float16* wout_h    = (_Float16*)take((size_t)kD * kD * 2);
  _Float16* wctrl_h   = (_Float16*)take((size_t)kN3 * kD * 2);
  _Float16* wff1_h    = (_Float16*)take((size_t)kF * kD * 2);
  _Float16* wff2_h    = (_Float16*)take((size_t)kD * kF * 2);
  _Float16* q_h       = (_Float16*)take((size_t)kR * kD * 2);
  _Float16* k_h       = (_Float16*)take((size_t)kR * kD * 2);
  _Float16* vt_h      = (_Float16*)take((size_t)kR * kD * 2);
  _Float16* actx_h    = (_Float16*)take((size_t)kR * kD * 2);  // attn ctx (pre out-proj)
  _Float16* ctxp_h    = (_Float16*)take((size_t)kR * kD * 2);  // context (post out-proj), f16
  _Float16* hn_h      = (_Float16*)take((size_t)kR * kD * 2);
  _Float16* h1_h      = (_Float16*)take((size_t)kR * kF * 2);
  float*    attn_out  = (float*)take((size_t)kR * kD * 4);     // context f32
  float*    x1        = (float*)take((size_t)kR * kD * 4);
  float*    xnorm1    = (float*)take((size_t)kR * kD * 4);
  float*    x2        = (float*)take((size_t)kR * kD * 4);
  float*    ctrl      = (float*)take((size_t)kR * kN3 * 4);

  // 1) weights fp32 -> f16
  auto cvt = [&](const float* src, _Float16* dst, int n) {
    f32_to_f16_kernel<<<(n + 255) / 256, 256, 0, stream>>>(src, dst, n);
  };
  cvt(in_proj_w,  wqkv_h,  kN3 * kD);
  cvt(out_proj_w, wout_h,  kD * kD);
  cvt(ctrl_w,     wctrl_h, kN3 * kD);
  cvt(w_ff1,      wff1_h,  kF * kD);
  cvt(w_ff2,      wff2_h,  kD * kF);

  // 2) xn = LN(x) -> f16
  layernorm_kernel<<<kR, 256, 0, stream>>>(x, g_attn, b_attn, nullptr, xn_h);

  // 3) QKV projection + head-major scatter (V transposed)
  gemm_kernel<1><<<dim3(kN3 / 64, kR / 64), 128, 0, stream>>>(
      xn_h, wqkv_h, in_proj_b, kR, kN3, kD,
      nullptr, q_h, nullptr, nullptr, k_h, vt_h);

  // 4) causal flash attention -> actx_h [row, h*64+d]
  flash_attn_kernel<<<dim3(kS / 64, kH, kB), 128, 0, stream>>>(q_h, k_h, vt_h,
                                                               actx_h);

  // 5) out-proj: context f32/f16 + x1 = x + attn_output
  gemm_kernel<2><<<dim3(kD / 64, kR / 64), 128, 0, stream>>>(
      actx_h, wout_h, out_proj_b, kR, kD, kD,
      attn_out, ctxp_h, x, x1, nullptr, nullptr);

  // 6) x_norm = LN(x1) -> f32
  layernorm_kernel<<<kR, 256, 0, stream>>>(x1, g1, b1, xnorm1, nullptr);

  // 7) ctrl = context @ ctrl_w^T + ctrl_b -> f32
  gemm_kernel<0><<<dim3(kN3 / 64, kR / 64), 128, 0, stream>>>(
      ctxp_h, wctrl_h, ctrl_b, kR, kN3, kD,
      ctrl, nullptr, nullptr, nullptr, nullptr, nullptr);

  // 8) INL loop: trajectories -> d_out, x2 = x1 + x_state
  inl_kernel<<<(kR * kD) / 256, 256, 0, stream>>>(
      xnorm1, ctrl, attn_out, mu_global, mu_offsets, x1, x2, trajx, trajv);

  // 9) h = LN(x2) -> f16
  layernorm_kernel<<<kR, 256, 0, stream>>>(x2, g2, b2, nullptr, hn_h);

  // 10) FF1 + GELU -> f16
  gemm_kernel<3><<<dim3(kF / 64, kR / 64), 128, 0, stream>>>(
      hn_h, wff1_h, b_ff1, kR, kF, kD,
      nullptr, h1_h, nullptr, nullptr, nullptr, nullptr);

  // 11) FF2 + residual -> d_out.x
  gemm_kernel<4><<<dim3(kD / 64, kR / 64), 128, 0, stream>>>(
      h1_h, wff2_h, b_ff2, kR, kD, kF,
      out_x, nullptr, x2, nullptr, nullptr, nullptr);
}
